// MOE_7799660610026
// MI455X (gfx1250) — matrix operand order repaired and checked by reading, not compiled, not run on hardware
//
#include <hip/hip_runtime.h>
#include <hip/hip_bf16.h>

#define NEXP 160
#define TOPK 6
#define SEQ  2560
#define HID  4096
#define ISZ  128
#define CAP  96   // TOPK*SEQ/NEXP

typedef __bf16 bf16x4  __attribute__((ext_vector_type(4)));
typedef __bf16 bf16x8  __attribute__((ext_vector_type(8)));
typedef __bf16 bf16x16 __attribute__((ext_vector_type(16)));
typedef float  f32x8   __attribute__((ext_vector_type(8)));

typedef __attribute__((address_space(3))) void lds_void;

// 32-bit LDS (workgroup-relative) offset of a __shared__ object, for async-to-LDS ops.
static __device__ __forceinline__ unsigned lds_addr(void* p) {
  return (unsigned)(unsigned long long)(lds_void*)p;
}

// Convert 16 contiguous fp32 (64B, 16B-aligned) into a bf16x16 fragment.
static __device__ __forceinline__ bf16x16 cvt16(const float* __restrict__ p) {
  float4 a = *(const float4*)(p + 0);
  float4 b = *(const float4*)(p + 4);
  float4 c = *(const float4*)(p + 8);
  float4 d = *(const float4*)(p + 12);
  bf16x16 r;
  r[0]  = (__bf16)a.x; r[1]  = (__bf16)a.y; r[2]  = (__bf16)a.z; r[3]  = (__bf16)a.w;
  r[4]  = (__bf16)b.x; r[5]  = (__bf16)b.y; r[6]  = (__bf16)b.z; r[7]  = (__bf16)b.w;
  r[8]  = (__bf16)c.x; r[9]  = (__bf16)c.y; r[10] = (__bf16)c.z; r[11] = (__bf16)c.w;
  r[12] = (__bf16)d.x; r[13] = (__bf16)d.y; r[14] = (__bf16)d.z; r[15] = (__bf16)d.w;
  return r;
}

static __device__ __forceinline__ bf16x16 lds_afrag(const __bf16* __restrict__ ap) {
  bf16x8 lo = *(const bf16x8*)ap;          // K = k0+g*16 .. +7
  bf16x8 hi = *(const bf16x8*)(ap + 8);    // K = k0+g*16+8 .. +15
  return __builtin_shufflevector(lo, hi, 0, 1, 2, 3, 4, 5, 6, 7,
                                 8, 9, 10, 11, 12, 13, 14, 15);
}

// ---------------- Kernel 0: invert the re_index permutation ----------------
__global__ void moe_invperm(const int* __restrict__ re, int* __restrict__ inv, int n) {
  int i = blockIdx.x * blockDim.x + threadIdx.x;
  if (i < n) inv[re[i]] = i;
}

// ---------------- Kernel 1: out = shared_out (float4 copy) ----------------
__global__ void moe_initout(const float* __restrict__ sh, float* __restrict__ out, int n4) {
  int i = blockIdx.x * blockDim.x + threadIdx.x;
  if (i < n4) ((float4*)out)[i] = ((const float4*)sh)[i];
}

// ---------------- Kernel 2: gathered gate/up GEMM + SiLU -> act (bf16) -----
// grid.x = expert; 256 threads = 8 waves; wave w owns N-tile w (16 cols) of
// BOTH gate and up, looping all 6 M-tiles. x chunk (96 x 64) staged in LDS as
// bf16, double-buffered through registers: chunk k+1 global loads are issued
// before the WMMA work on chunk k so HBM latency overlaps the matrix core.
__global__ __launch_bounds__(256) void moe_gateup(
    const float* __restrict__ hidden, const float* __restrict__ Wg,
    const float* __restrict__ Wu, const int* __restrict__ tokidx,
    __bf16* __restrict__ act) {
  const int e    = blockIdx.x;
  const int tid  = threadIdx.x;
  const int wv   = tid >> 5;
  const int lane = tid & 31;
  const int lg   = lane >> 4;   // half-wave group (A/B K-split)
  const int ln   = lane & 15;

  __shared__ __align__(16) __bf16 xs[96 * 72];   // rows padded to 72 bf16 (144B)
  __shared__ int tok[96];
  if (tid < 96) tok[tid] = tokidx[e * CAP + tid];
  __syncthreads();

  // Loop-invariant staging geometry: thread handles 6 float4 quads per chunk.
  const float* rowp[6];
  int          soff[6];
#pragma unroll
  for (int i = 0; i < 6; ++i) {
    int q   = tid + 256 * i;
    int row = q >> 4;
    int qi  = q & 15;
    rowp[i] = hidden + (size_t)tok[row] * HID + qi * 4;
    soff[i] = row * 72 + qi * 4;
  }

  f32x8 accg[6], accu[6];
#pragma unroll
  for (int mt = 0; mt < 6; ++mt) {
    accg[mt] = (f32x8){0.f, 0.f, 0.f, 0.f, 0.f, 0.f, 0.f, 0.f};
    accu[mt] = (f32x8){0.f, 0.f, 0.f, 0.f, 0.f, 0.f, 0.f, 0.f};
  }

  const float* wgRow = Wg + (size_t)e * ISZ * HID + (size_t)(wv * 16 + ln) * HID;
  const float* wuRow = Wu + (size_t)e * ISZ * HID + (size_t)(wv * 16 + ln) * HID;

  float4 pf[6];
#pragma unroll
  for (int i = 0; i < 6; ++i) pf[i] = *(const float4*)(rowp[i]);  // prefetch chunk 0

  for (int kc = 0; kc < HID; kc += 64) {
    __syncthreads();  // all waves done reading xs (previous chunk)
#pragma unroll
    for (int i = 0; i < 6; ++i) {
      bf16x4 pk = {(__bf16)pf[i].x, (__bf16)pf[i].y, (__bf16)pf[i].z, (__bf16)pf[i].w};
      *(bf16x4*)&xs[soff[i]] = pk;
    }
    __syncthreads();  // xs for this chunk ready

    if (kc + 64 < HID) {  // issue next chunk's gathered loads now
#pragma unroll
      for (int i = 0; i < 6; ++i) pf[i] = *(const float4*)(rowp[i] + kc + 64);
    }

#pragma unroll
    for (int s = 0; s < 2; ++s) {            // two K=32 steps per chunk
      int cb = kc + s * 32 + lg * 16;        // B frag: 16 contiguous K fp32
      bf16x16 bg = cvt16(wgRow + cb);
      bf16x16 bu = cvt16(wuRow + cb);
#pragma unroll
      for (int mt = 0; mt < 6; ++mt) {
        const __bf16* ap = &xs[(mt * 16 + ln) * 72 + s * 32 + lg * 16];
        bf16x16 a = lds_afrag(ap);
        accg[mt] = __builtin_amdgcn_wmma_f32_16x16x32_bf16(
            false, a, false, bg, (short)0, accg[mt], false, false);
        accu[mt] = __builtin_amdgcn_wmma_f32_16x16x32_bf16(
            false, a, false, bu, (short)0, accu[mt], false, false);
      }
    }
  }

  // Epilogue: act = silu(g) * u, stored bf16 row-major [CAP][ISZ].
  __bf16* ap = act + (size_t)e * CAP * ISZ;
#pragma unroll
  for (int mt = 0; mt < 6; ++mt) {
#pragma unroll
    for (int v = 0; v < 8; ++v) {
      float g  = accg[mt][v];
      float u  = accu[mt][v];
      float sg = g / (1.f + __expf(-g));
      int row  = mt * 16 + v + lg * 8;
      ap[row * ISZ + wv * 16 + ln] = (__bf16)(sg * u);
    }
  }
}

// ---------------- Kernel 3: down GEMM + weighted atomic scatter ------------
// grid = (expert, nsplit 0..7); each block covers 512 of the 4096 H columns.
// The 24KB bf16 act tile is staged with CDNA5 async global->LDS DMA
// (GLOBAL_LOAD_ASYNC_TO_LDS_B128, tracked on ASYNCcnt) — no VGPR round-trip.
__global__ __launch_bounds__(256) void moe_down(
    const __bf16* __restrict__ act, const float* __restrict__ Wd,
    const int* __restrict__ inv, const float* __restrict__ topw,
    float* __restrict__ out) {
  const int e    = blockIdx.x;
  const int ns   = blockIdx.y;
  const int tid  = threadIdx.x;
  const int wv   = tid >> 5;
  const int lane = tid & 31;
  const int lg   = lane >> 4;
  const int ln   = lane & 15;

  __shared__ __align__(16) __bf16 as[96 * 128];  // 24KB act tile
  __shared__ float  wrow[96];
  __shared__ int    trow[96];

  // Async-stage the act tile: 1536 x 16B chunks, 6 per thread.
  {
    const unsigned asbase = lds_addr(as);
    const __bf16* ag = act + (size_t)e * CAP * ISZ;
#pragma unroll
    for (int i = 0; i < 6; ++i) {
      int c = tid + 256 * i;
      unsigned long long gp = (unsigned long long)(ag + c * 8);
      unsigned la = asbase + c * 16;
      asm volatile("global_load_async_to_lds_b128 %0, %1, off"
                   :: "v"(la), "v"(gp) : "memory");
    }
  }

  if (tid < 96) {
    int j = inv[e * CAP + tid];     // position in (topk, token) order
    int k = j / SEQ;
    int t = j - k * SEQ;
    wrow[tid] = topw[k * SEQ + t];  // topk_weight[k, t, 0]
    trow[tid] = t;
  }
  asm volatile("s_wait_asynccnt 0x0" ::: "memory");
  __syncthreads();

#pragma unroll
  for (int q = 0; q < 4; ++q) {
    int nt = ns * 32 + wv * 4 + q;  // global 16-col tile of H
    int h  = nt * 16 + ln;
    const float* wd = Wd + (size_t)e * HID * ISZ + (size_t)h * ISZ;
    bf16x16 bfr[4];
#pragma unroll
    for (int ks = 0; ks < 4; ++ks) bfr[ks] = cvt16(wd + ks * 32 + lg * 16);
#pragma unroll
    for (int mt = 0; mt < 6; ++mt) {
      f32x8 acc = (f32x8){0.f, 0.f, 0.f, 0.f, 0.f, 0.f, 0.f, 0.f};
#pragma unroll
      for (int ks = 0; ks < 4; ++ks) {
        const __bf16* apt = &as[(mt * 16 + ln) * 128 + ks * 32 + lg * 16];
        bf16x16 a = lds_afrag(apt);
        acc = __builtin_amdgcn_wmma_f32_16x16x32_bf16(
            false, a, false, bfr[ks], (short)0, acc, false, false);
      }
#pragma unroll
      for (int v = 0; v < 8; ++v) {
        int row = mt * 16 + v + lg * 8;
        atomicAdd(out + (size_t)trow[row] * HID + h, wrow[row] * acc[v]);
      }
    }
  }
}

extern "C" void kernel_launch(void* const* d_in, const int* in_sizes, int n_in,
                              void* d_out, int out_size, void* d_ws, size_t ws_size,
                              hipStream_t stream) {
  const float* hidden = (const float*)d_in[0];
  const float* topw   = (const float*)d_in[1];
  const float* shared = (const float*)d_in[2];
  const float* Wg     = (const float*)d_in[3];
  const float* Wu     = (const float*)d_in[4];
  const float* Wd     = (const float*)d_in[5];
  const int*   tokidx = (const int*)d_in[6];
  const int*   re     = (const int*)d_in[7];
  float* out = (float*)d_out;

  int*    inv = (int*)d_ws;                           // 160*96 ints
  __bf16* act = (__bf16*)((char*)d_ws + 262144);      // 160*96*128 bf16

  const int nperm = NEXP * CAP;  // 15360
  moe_invperm<<<(nperm + 255) / 256, 256, 0, stream>>>(re, inv, nperm);

  const int n4 = SEQ * HID / 4;  // 2,621,440 float4
  moe_initout<<<(n4 + 255) / 256, 256, 0, stream>>>(shared, out, n4);

  moe_gateup<<<NEXP, 256, 0, stream>>>(hidden, Wg, Wu, tokidx, act);

  moe_down<<<dim3(NEXP, 8), 256, 0, stream>>>(act, Wd, inv, topw, out);
}
